// SE_Fusion_Four3D_19292993093897
// MI455X (gfx1250) — matrix operand order repaired and checked by reading, not compiled, not run on hardware
//
#include <hip/hip_runtime.h>
#include <hip/hip_bf16.h>

typedef __attribute__((ext_vector_type(16))) __bf16 v16bf;
typedef __attribute__((ext_vector_type(8)))  float  v8f;

#define LDSW   520        // padded bf16 row stride (512 cols + 8) -> no 64-bank conflicts
#define MAT_DW 131072     // dwords per packed 512x512 bf16 weight in frag layout
#define ROWS   64         // rows per block (4 M-tiles of 16)

__device__ __forceinline__ unsigned short f2bf(float f) {
  unsigned int u = __float_as_uint(f);
  u += 0x7FFFu + ((u >> 16) & 1u);   // round-to-nearest-even
  return (unsigned short)(u >> 16);
}

// A-frag k order used here (contiguous, not AMD's table order): VGPR v holds K pair starting at kbase(v, lane>>4) = 16*half + 2v (matches B pack)
__device__ __forceinline__ int kbase(int v, int half) {
  return (half << 4) + (v << 1);
}

// ---------------------------------------------------------------------------
// Kernel 1: convert weights f32 -> bf16 and pack into WMMA B-fragment layout.
//   ws dword index = mat*MAT_DW + ((kt*32 + nt)*32 + lane)*8 + v
//   B-frag (32x16): lane holds column n = lane&15, K half = (lane>>4)*16,
//   VGPR v packs K = {base+2v, base+2v+1}.
//   mat 0   : W_gcn (512x512)
//   mat 1..4: W1sum_i = sum over the 4 K-blocks of W1_i (2048x512 -> 512x512)
//   mat 5..8: W2_i (512x512)
// ---------------------------------------------------------------------------
__global__ void pack_weights(const float* __restrict__ Wg,
                             const float* __restrict__ W1a, const float* __restrict__ W1b,
                             const float* __restrict__ W1c, const float* __restrict__ W1d,
                             const float* __restrict__ W2a, const float* __restrict__ W2b,
                             const float* __restrict__ W2c, const float* __restrict__ W2d,
                             unsigned int* __restrict__ ws) {
  int gid = blockIdx.x * blockDim.x + threadIdx.x;
  if (gid >= 9 * MAT_DW) return;
  int mat  = gid / MAT_DW;
  int rem  = gid - mat * MAT_DW;
  int v    = rem & 7;
  int lane = (rem >> 3) & 31;
  int t    = rem >> 8;          // kt*32 + nt
  int nt   = t & 31;
  int kt   = t >> 5;
  int n    = nt * 16 + (lane & 15);
  int k0   = kt * 32 + (lane >> 4) * 16 + v * 2;

  float e0, e1;
  if (mat == 0) {
    e0 = Wg[(size_t)k0 * 512 + n];
    e1 = Wg[(size_t)(k0 + 1) * 512 + n];
  } else if (mat <= 4) {
    const float* W1 = (mat == 1) ? W1a : (mat == 2) ? W1b : (mat == 3) ? W1c : W1d;
    e0 = 0.f; e1 = 0.f;
#pragma unroll
    for (int m = 0; m < 4; ++m) {
      e0 += W1[(size_t)(m * 512 + k0) * 512 + n];
      e1 += W1[(size_t)(m * 512 + k0 + 1) * 512 + n];
    }
  } else {
    const float* W2 = (mat == 5) ? W2a : (mat == 6) ? W2b : (mat == 7) ? W2c : W2d;
    e0 = W2[(size_t)k0 * 512 + n];
    e1 = W2[(size_t)(k0 + 1) * 512 + n];
  }
  ws[gid] = (unsigned int)f2bf(e0) | ((unsigned int)f2bf(e1) << 16);
}

// ---------------------------------------------------------------------------
// 64x512 @ 512x512 tile GEMM over K=512 (16 k-tiles of 32), bf16 WMMA.
// Each wave: 4 M-tiles x 4 N-tiles => 16 C-frags; each A-frag and each B-frag
// feeds 4 WMMAs (double the reuse of a 2x4 blocking -> half the L2/LDS bytes
// per matrix op).
// ---------------------------------------------------------------------------
__device__ __forceinline__ void mma_512(const unsigned short* __restrict__ aLds,
                                        const unsigned int*   __restrict__ wB,
                                        int nt0, int lane, v8f acc[4][4]) {
  const v8f zero = {0.f, 0.f, 0.f, 0.f, 0.f, 0.f, 0.f, 0.f};
#pragma unroll
  for (int mt = 0; mt < 4; ++mt)
#pragma unroll
    for (int j = 0; j < 4; ++j) acc[mt][j] = zero;

  const int half = lane >> 4;
  const int mrow = lane & 15;

  for (int kt = 0; kt < 16; ++kt) {
    union { v16bf v; unsigned int u[8]; } a[4];
#pragma unroll
    for (int mt = 0; mt < 4; ++mt) {
      const unsigned short* rp = aLds + (mt * 16 + mrow) * LDSW + kt * 32;
#pragma unroll
      for (int v = 0; v < 8; ++v)
        a[mt].u[v] = *(const unsigned int*)(rp + kbase(v, half));   // ds_load
    }
#pragma unroll
    for (int j = 0; j < 4; ++j) {
      union { v16bf v; uint4 q[2]; } b;
      const uint4* bp = (const uint4*)(wB + (((size_t)(kt * 32 + nt0 + j) * 32 + lane) << 3));
      b.q[0] = bp[0];                               // global_load_b128 x2
      b.q[1] = bp[1];
#pragma unroll
      for (int mt = 0; mt < 4; ++mt)
        acc[mt][j] = __builtin_amdgcn_wmma_f32_16x16x32_bf16(
            false, a[mt].v, false, b.v, (short)0, acc[mt][j], false, false);
    }
  }
}

// ---------------------------------------------------------------------------
// Fused kernel: 64 rows of (B*D)=32768 per block; 8 wave32s (256 threads).
//   fmix = 0.25*(fea1+..+fea4)            -> LDS bf16
//   u    = fmix @ Wg + b_gcn              -> LDS bf16     (== every z block)
//   per branch i:
//     h  = relu(u @ W1sum_i + b1_i)       -> LDS bf16
//     g  = sigmoid(h @ W2_i + b2_i)
//     out_i = g * fea_i                   -> global f32
// ---------------------------------------------------------------------------
__global__ void __launch_bounds__(256)
fused_main(const float* __restrict__ fea1, const float* __restrict__ fea2,
           const float* __restrict__ fea3, const float* __restrict__ fea4,
           const float* __restrict__ b_gcn,
           const float* __restrict__ b1_1, const float* __restrict__ b1_2,
           const float* __restrict__ b1_3, const float* __restrict__ b1_4,
           const float* __restrict__ b2_1, const float* __restrict__ b2_2,
           const float* __restrict__ b2_3, const float* __restrict__ b2_4,
           const unsigned int* __restrict__ wfrag,
           float* __restrict__ out) {
  extern __shared__ unsigned short smem[];
  unsigned short* ldsMix = smem;                 // [ROWS][LDSW]
  unsigned short* ldsU   = smem + ROWS * LDSW;   // [ROWS][LDSW]
  unsigned short* ldsH   = smem + 2 * ROWS * LDSW;

  const int tid  = threadIdx.x;
  const int lane = tid & 31;
  const int wave = tid >> 5;
  const int nt0  = wave * 4;                     // 8 waves x 4 n-tiles = 512 cols
  const int row0 = blockIdx.x * ROWS;

  // ---- Stage 0: feature mix -> bf16 LDS (A-operand of GCN GEMM) ----
  for (int p = tid; p < ROWS * 256; p += 256) {
    int r  = p >> 8;
    int c2 = (p & 255) * 2;
    size_t g = (size_t)(row0 + r) * 512 + c2;
    float2 a = *(const float2*)(fea1 + g);
    float2 b = *(const float2*)(fea2 + g);
    float2 c = *(const float2*)(fea3 + g);
    float2 d = *(const float2*)(fea4 + g);
    float x0 = 0.25f * (a.x + b.x + c.x + d.x);
    float x1 = 0.25f * (a.y + b.y + c.y + d.y);
    *(unsigned int*)(ldsMix + r * LDSW + c2) =
        (unsigned int)f2bf(x0) | ((unsigned int)f2bf(x1) << 16);
  }
  __syncthreads();

  // ---- Stage 1: u = fmix @ W_gcn + b_gcn ----
  {
    v8f acc[4][4];
    mma_512(ldsMix, wfrag, nt0, lane, acc);
#pragma unroll
    for (int j = 0; j < 4; ++j) {
      int col = (nt0 + j) * 16 + (lane & 15);
      float bias = b_gcn[col];
#pragma unroll
      for (int mt = 0; mt < 4; ++mt)
#pragma unroll
        for (int e = 0; e < 8; ++e) {
          int M = mt * 16 + (lane >> 4) * 8 + e;   // C-frag: VGPR e -> row
          ldsU[M * LDSW + col] = f2bf(acc[mt][j][e] + bias);
        }
    }
  }
  __syncthreads();

  // ---- Stage 2: four SE gates ----
  const float* feas[4] = {fea1, fea2, fea3, fea4};
  const float* b1s[4]  = {b1_1, b1_2, b1_3, b1_4};
  const float* b2s[4]  = {b2_1, b2_2, b2_3, b2_4};

#pragma unroll
  for (int i = 0; i < 4; ++i) {
    { // h = relu(u @ W1sum_i + b1_i)
      v8f acc[4][4];
      mma_512(ldsU, wfrag + (size_t)(1 + i) * MAT_DW, nt0, lane, acc);
#pragma unroll
      for (int j = 0; j < 4; ++j) {
        int col = (nt0 + j) * 16 + (lane & 15);
        float bias = b1s[i][col];
#pragma unroll
        for (int mt = 0; mt < 4; ++mt)
#pragma unroll
          for (int e = 0; e < 8; ++e) {
            int M = mt * 16 + (lane >> 4) * 8 + e;
            ldsH[M * LDSW + col] = f2bf(fmaxf(acc[mt][j][e] + bias, 0.0f));
          }
      }
    }
    __syncthreads();

    { // out_i = sigmoid(h @ W2_i + b2_i) * fea_i
      v8f acc[4][4];
      mma_512(ldsH, wfrag + (size_t)(5 + i) * MAT_DW, nt0, lane, acc);
      const float* fi = feas[i];
      float* oi = out + (size_t)i * (64ull * 512 * 512);
#pragma unroll
      for (int j = 0; j < 4; ++j) {
        int col = (nt0 + j) * 16 + (lane & 15);
        float bias = b2s[i][col];
#pragma unroll
        for (int mt = 0; mt < 4; ++mt)
#pragma unroll
          for (int e = 0; e < 8; ++e) {
            int M = mt * 16 + (lane >> 4) * 8 + e;
            size_t g = (size_t)(row0 + M) * 512 + col;
            float x = acc[mt][j][e] + bias;
            float gate = 1.0f / (1.0f + __expf(-x));   // relu(sigmoid(x)) == sigmoid(x)
            oi[g] = gate * fi[g];
          }
      }
    }
    __syncthreads();
  }
}

// ---------------------------------------------------------------------------
extern "C" void kernel_launch(void* const* d_in, const int* in_sizes, int n_in,
                              void* d_out, int out_size, void* d_ws, size_t ws_size,
                              hipStream_t stream) {
  const float* fea1  = (const float*)d_in[0];
  const float* fea2  = (const float*)d_in[1];
  const float* fea3  = (const float*)d_in[2];
  const float* fea4  = (const float*)d_in[3];
  const float* W_gcn = (const float*)d_in[4];
  const float* b_gcn = (const float*)d_in[5];
  const float* W1[4] = {(const float*)d_in[6],  (const float*)d_in[10],
                        (const float*)d_in[14], (const float*)d_in[18]};
  const float* b1[4] = {(const float*)d_in[7],  (const float*)d_in[11],
                        (const float*)d_in[15], (const float*)d_in[19]};
  const float* W2[4] = {(const float*)d_in[8],  (const float*)d_in[12],
                        (const float*)d_in[16], (const float*)d_in[20]};
  const float* b2[4] = {(const float*)d_in[9],  (const float*)d_in[13],
                        (const float*)d_in[17], (const float*)d_in[21]};
  unsigned int* ws = (unsigned int*)d_ws;
  float* out = (float*)d_out;

  // 9 packed 512x512 bf16 matrices = 4.5 MiB in d_ws (L2-resident thereafter)
  {
    int total = 9 * MAT_DW;
    pack_weights<<<dim3((total + 255) / 256), dim3(256), 0, stream>>>(
        W_gcn, W1[0], W1[1], W1[2], W1[3], W2[0], W2[1], W2[2], W2[3], ws);
  }

  // 32768 rows / 64 rows per block
  size_t smem_bytes = (size_t)3 * ROWS * LDSW * sizeof(unsigned short);  // 199,680 B
  fused_main<<<dim3(32768 / ROWS), dim3(256), smem_bytes, stream>>>(
      fea1, fea2, fea3, fea4, b_gcn,
      b1[0], b1[1], b1[2], b1[3],
      b2[0], b2[1], b2[2], b2[3],
      ws, out);
}
